// MAB_16965120819751
// MI455X (gfx1250) — matrix operand order repaired and checked by reading, not compiled, not run on hardware
//
#include <hip/hip_runtime.h>
#include <hip/hip_bf16.h>
#include <stdint.h>

// ---------------------------------------------------------------------------
// Problem constants (from reference): B=8, SQ=SK=1024, D=1024, H=16, dh=64
// ---------------------------------------------------------------------------
#define BB   8
#define SQL  1024
#define SKL  1024
#define DVL  1024
#define HH   16
#define DH   64
#define MROWS (BB * SQL)   // 8192

typedef __attribute__((ext_vector_type(16))) __bf16 v16bf;
typedef __attribute__((ext_vector_type(8)))  __bf16 bf16x8;
typedef __attribute__((ext_vector_type(8)))  float  v8f;
typedef __attribute__((ext_vector_type(4)))  float  f32x4;

// Build a wave32 WMMA fragment (16 bf16/lane) from two contiguous 8x bf16 runs
// (the documented A/B VGPR layouts are always two contiguous 8-element runs).
__device__ __forceinline__ v16bf frag_from2(const __bf16* p0, const __bf16* p1) {
    bf16x8 lo = *(const bf16x8*)p0;
    bf16x8 hi = *(const bf16x8*)p1;
    v16bf r;
#pragma unroll
    for (int i = 0; i < 8; ++i) { r[i] = lo[i]; r[i + 8] = hi[i]; }
    return r;
}

__device__ __forceinline__ v8f wmma_bf16(v16bf a, v16bf b, v8f c) {
    return __builtin_amdgcn_wmma_f32_16x16x32_bf16(
        false, a, false, b, (short)0, c, false, false);
}

// Raw v_max_num_f32 (already IEEE maximumNumber) -- avoids the canonicalize
// ops clang inserts around fmaxf.
__device__ __forceinline__ float fmaxr(float a, float b) {
    float r;
    asm("v_max_num_f32 %0, %1, %2" : "=v"(r) : "v"(a), "v"(b));
    return r;
}

// ---- VALU lane-group butterfly reductions (v_permlane16/x16_b32) -----------
__device__ __forceinline__ float permx16(float v, int s0, int s1) {
    int i = __builtin_bit_cast(int, v);
    int r = __builtin_amdgcn_permlane16(i, i, s0, s1, false, false);
    return __builtin_bit_cast(float, r);
}
__device__ __forceinline__ float permxx16(float v) {   // xor-16 across groups
    int i = __builtin_bit_cast(int, v);
    int r = __builtin_amdgcn_permlanex16(i, i, 0x76543210, 0xFEDCBA98, false, false);
    return __builtin_bit_cast(float, r);
}
__device__ __forceinline__ float rmax16(float v) {
    v = fmaxr(v, permx16(v, 0x67452301, 0xEFCDAB89));   // xor 1
    v = fmaxr(v, permx16(v, 0x54761032, 0xDCFE98BA));   // xor 2
    v = fmaxr(v, permx16(v, 0x32107654, 0xBA98FEDC));   // xor 4
    v = fmaxr(v, permx16(v, 0xFEDCBA98, 0x76543210));   // xor 8
    return v;
}
__device__ __forceinline__ float rsum16(float v) {
    v += permx16(v, 0x67452301, 0xEFCDAB89);
    v += permx16(v, 0x54761032, 0xDCFE98BA);
    v += permx16(v, 0x32107654, 0xBA98FEDC);
    v += permx16(v, 0xFEDCBA98, 0x76543210);
    return v;
}
__device__ __forceinline__ float wave_sum32(float v) {
    v = rsum16(v);
    return v + permxx16(v);
}

// ---- CDNA5 async global->LDS copy (ASYNCcnt) -------------------------------
// Generic shared pointers carry the LDS byte offset in their low 32 bits
// (flat LDS aperture).
__device__ __forceinline__ unsigned lds_addr32(const void* p) {
    return (unsigned)(uintptr_t)p;
}
__device__ __forceinline__ void async_copy_b128(unsigned lds, const void* g) {
    asm volatile("global_load_async_to_lds_b128 %0, %1, off"
                 :: "v"(lds), "v"((unsigned long long)(uintptr_t)g) : "memory");
}
__device__ __forceinline__ void wait_async0() {
    asm volatile("s_wait_asynccnt 0" ::: "memory");
}

// ---------------------------------------------------------------------------
// cvt_bf16_kernel: flat f32 -> bf16 convert (Q, K inputs). 16 elems/thread.
// ---------------------------------------------------------------------------
__global__ __launch_bounds__(256) void cvt_bf16_kernel(
    const float* __restrict__ X, __bf16* __restrict__ Y)
{
    const size_t i = ((size_t)blockIdx.x * 256 + threadIdx.x) * 16;
    f32x4 a[4];
#pragma unroll
    for (int q = 0; q < 4; ++q) a[q] = *(const f32x4*)(X + i + 4 * q);
    bf16x8 o0, o1;
#pragma unroll
    for (int e = 0; e < 4; ++e) {
        o0[e] = (__bf16)a[0][e]; o0[e + 4] = (__bf16)a[1][e];
        o1[e] = (__bf16)a[2][e]; o1[e + 4] = (__bf16)a[3][e];
    }
    *(bf16x8*)(Y + i) = o0;
    *(bf16x8*)(Y + i + 8) = o1;
}

// ---------------------------------------------------------------------------
// transpose_w_kernel: W[k][n] f32 -> Wt[n][k] bf16, 64x64 LDS tiles.
// ---------------------------------------------------------------------------
__global__ __launch_bounds__(256) void transpose_w_kernel(
    const float* __restrict__ W, __bf16* __restrict__ Wt)
{
    __shared__ __align__(16) __bf16 lt[64][72];
    const int tid = threadIdx.x;
    const int kt = blockIdx.y * 64, nt = blockIdx.x * 64;
    const int r = tid >> 2, c0 = (tid & 3) * 16;

    const float* s = W + (size_t)(kt + r) * DVL + nt + c0;
#pragma unroll
    for (int j = 0; j < 16; ++j) lt[r][c0 + j] = (__bf16)s[j];
    __syncthreads();

    bf16x8 o0, o1;
#pragma unroll
    for (int j = 0; j < 8; ++j) { o0[j] = lt[c0 + j][r]; o1[j] = lt[c0 + 8 + j][r]; }
    __bf16* d = Wt + (size_t)(nt + r) * DVL + kt + c0;
    *(bf16x8*)d = o0;
    *(bf16x8*)(d + 8) = o1;
}

// ---------------------------------------------------------------------------
// proj_kernel: out = epilogue( A[M,1024](bf16) @ W + bias ),  W given as
// pre-transposed bf16 Wt[n][k].  Both tiles staged with async global->LDS.
//   EPI==0 : outF = acc+bias (opt);  outB = bf16((acc+bias)*oscale) (opt),
//            TRANSB ? outB laid out [B,H,dh,S] (packed b128 stores) : [M,N]
//   EPI==1 : outF = resid + relu(acc+bias)
// Block 256 (8 waves), tile 128x128, k-step 64, double-buffered (one barrier
// per 64-deep k-step; 16 WMMA per wave per step).
// ---------------------------------------------------------------------------
template <int EPI, bool TRANSB>
__global__ __launch_bounds__(256) void proj_kernel(
    const __bf16* __restrict__ A,
    const __bf16* __restrict__ Wt,
    const float*  __restrict__ bias,
    const __bf16* __restrict__ resid,
    float*  __restrict__ outF,
    __bf16* __restrict__ outB,
    float oscale)
{
    __shared__ __align__(16) __bf16 lA[2][128][72];   // A tile  [m][k0..63]
    __shared__ __align__(16) __bf16 lBt[2][128][72];  // Wt tile [n][k0..63]

    const int tid  = threadIdx.x;
    const int w    = tid >> 5;
    const int lane = tid & 31;
    const int half = lane >> 4;
    const int l16  = lane & 15;
    const int waveM = w >> 1;
    const int waveN = w & 1;
    const int mtile = blockIdx.y * 128;
    const int ntile = blockIdx.x * 128;

    // staging: 2 threads per row, 32 elems each, for both tiles
    const int srow = tid >> 1, sc0 = (tid & 1) * 32;
    const __bf16* gA = A  + (size_t)(mtile + srow) * DVL + sc0;
    const __bf16* gB = Wt + (size_t)(ntile + srow) * DVL + sc0;

    auto stage = [&](int buf, int kb) {
#pragma unroll
        for (int q = 0; q < 4; ++q) {
            async_copy_b128(lds_addr32(&lA[buf][srow][sc0 + 8 * q]),  gA + kb + 8 * q);
            async_copy_b128(lds_addr32(&lBt[buf][srow][sc0 + 8 * q]), gB + kb + 8 * q);
        }
    };

    stage(0, 0);
    wait_async0();
    __syncthreads();

    v8f acc[2][4];
#pragma unroll
    for (int g = 0; g < 2; ++g)
#pragma unroll
        for (int t = 0; t < 4; ++t) acc[g][t] = v8f{0.f,0.f,0.f,0.f,0.f,0.f,0.f,0.f};

    for (int it = 0; it < DVL / 64; ++it) {
        const int cur = it & 1;
        const int kbn = (it + 1) * 64;
        if (kbn < DVL) stage(cur ^ 1, kbn);

#pragma unroll
        for (int c = 0; c < 2; ++c) {
            v16bf afr[2], bfr[4];
#pragma unroll
            for (int g = 0; g < 2; ++g) {
                const __bf16* p = &lA[cur][waveM * 32 + g * 16 + l16][c * 32 + half * 16];
                afr[g] = frag_from2(p, p + 8);
            }
#pragma unroll
            for (int t = 0; t < 4; ++t) {
                const __bf16* p = &lBt[cur][waveN * 64 + t * 16 + l16][c * 32 + half * 16];
                bfr[t] = frag_from2(p, p + 8);
            }
#pragma unroll
            for (int t = 0; t < 4; ++t)
#pragma unroll
                for (int g = 0; g < 2; ++g) acc[g][t] = wmma_bf16(afr[g], bfr[t], acc[g][t]);
        }

        wait_async0();
        __syncthreads();
    }

    // epilogue (C layout: elem j -> row j + 8*half, col = lane%16)
#pragma unroll
    for (int g = 0; g < 2; ++g)
#pragma unroll
        for (int t = 0; t < 4; ++t) {
            const int col = ntile + waveN * 64 + t * 16 + l16;
            const float bv = bias[col];
            if (EPI == 0 && TRANSB) {
                // packed transposed store: 8 consecutive rows are contiguous
                // along S in the [B,H,dh,S] layout
                bf16x8 pk;
#pragma unroll
                for (int j = 0; j < 8; ++j) pk[j] = (__bf16)((acc[g][t][j] + bv) * oscale);
                const int row0 = mtile + waveM * 32 + g * 16 + 8 * half;
                const size_t tidx =
                    (((size_t)(row0 >> 10) * HH + (col >> 6)) * DH + (col & 63)) * SKL
                    + (row0 & 1023);
                *(bf16x8*)(outB + tidx) = pk;
            } else {
#pragma unroll
                for (int j = 0; j < 8; ++j) {
                    const int row = mtile + waveM * 32 + g * 16 + j + 8 * half;
                    const size_t idx = (size_t)row * DVL + col;
                    float val = acc[g][t][j] + bv;
                    if (EPI == 0) {
                        if (outF) outF[idx] = val;
                        if (outB) outB[idx] = (__bf16)(val * oscale);
                    } else {
                        float r = (float)resid[idx];
                        outF[idx] = r + (val > 0.f ? val : 0.f);
                    }
                }
            }
        }
}

// ---------------------------------------------------------------------------
// attn_kernel: flash attention per (b, h, 128 q-rows); 8 waves x 16 rows.
// Q pre-scaled by 1/sqrt(1024); V given pre-transposed [B,H,dh,SK].
// 64-key stages, fully async double-buffered staging; 16 WMMA/wave/stage.
// ---------------------------------------------------------------------------
__global__ __launch_bounds__(256) void attn_kernel(
    const __bf16* __restrict__ q,
    const __bf16* __restrict__ k,
    const __bf16* __restrict__ vt,
    float* __restrict__ attn)
{
    __shared__ __align__(16) __bf16 lK[2][64][72];    // keys x dh
    __shared__ __align__(16) __bf16 lVt[2][64][72];   // dh x keys
    __shared__ __align__(16) __bf16 lP[8][16][72];    // per-wave P scratch

    const int tid  = threadIdx.x;
    const int w    = tid >> 5;
    const int lane = tid & 31;
    const int half = lane >> 4;
    const int l16  = lane & 15;
    const int h  = blockIdx.y;
    const int b  = blockIdx.z;
    const int q0 = blockIdx.x * 128 + w * 16;

    // staging: 4 threads per row, 16 elems each; same pattern for K and V^T
    const int kk = tid >> 2, c0 = (tid & 3) * 16;
    const __bf16* gK = k  + (size_t)(b * SKL + kk) * DVL + h * DH + c0;
    const __bf16* gV = vt + ((size_t)(b * HH + h) * DH + kk) * SKL + c0;

    auto stage = [&](int buf, int kbv) {
        const __bf16* pk = gK + (size_t)kbv * DVL;
        const __bf16* pv = gV + kbv;
        async_copy_b128(lds_addr32(&lK[buf][kk][c0]),      pk);
        async_copy_b128(lds_addr32(&lK[buf][kk][c0 + 8]),  pk + 8);
        async_copy_b128(lds_addr32(&lVt[buf][kk][c0]),     pv);
        async_copy_b128(lds_addr32(&lVt[buf][kk][c0 + 8]), pv + 8);
    };

    // Q fragments (A layout), kept in registers for the whole SK loop
    v16bf aQ[2];
    {
        const size_t qbase = ((size_t)(b * SQL + q0 + l16)) * DVL + h * DH;
#pragma unroll
        for (int c = 0; c < 2; ++c) {
            const __bf16* p0 = q + qbase + c * 32 + half * 16;
            aQ[c] = frag_from2(p0, p0 + 8);
        }
    }

    v8f o[4];
#pragma unroll
    for (int t = 0; t < 4; ++t) o[t] = v8f{0.f,0.f,0.f,0.f,0.f,0.f,0.f,0.f};
    float m[8], lsum[8];
#pragma unroll
    for (int j = 0; j < 8; ++j) { m[j] = -3.0e38f; lsum[j] = 0.f; }

    stage(0, 0);
    wait_async0();
    __syncthreads();

    for (int it = 0; it < SKL / 64; ++it) {
        const int cur = it & 1;
        if ((it + 1) < SKL / 64) stage(cur ^ 1, (it + 1) * 64);

        // scores: S[16 x 64] as four 16x16 C tiles, K-depth 64 (2 chunks)
        v8f s[4];
#pragma unroll
        for (int t = 0; t < 4; ++t) {
            const __bf16* p = &lK[cur][t * 16 + l16][half * 16];
            v16bf f0 = frag_from2(p, p + 8);
            v16bf f1 = frag_from2(p + 32, p + 40);
            v8f acc = v8f{0.f,0.f,0.f,0.f,0.f,0.f,0.f,0.f};
            acc = wmma_bf16(aQ[0], f0, acc);
            acc = wmma_bf16(aQ[1], f1, acc);
            s[t] = acc;
        }

        // online softmax over 64 keys (one butterfly pass per row)
        float rs[8];
#pragma unroll
        for (int j = 0; j < 8; ++j) {
            float mc = fmaxr(fmaxr(s[0][j], s[1][j]), fmaxr(s[2][j], s[3][j]));
            mc = rmax16(mc);
            float mn = fmaxr(m[j], mc);
            float f  = __expf(m[j] - mn);
            m[j] = mn;
            lsum[j] *= f;
#pragma unroll
            for (int t = 0; t < 4; ++t) o[t][j] *= f;
            float a = 0.f;
#pragma unroll
            for (int t = 0; t < 4; ++t) { s[t][j] = __expf(s[t][j] - mn); a += s[t][j]; }
            rs[j] = a;
        }
#pragma unroll
        for (int j = 0; j < 8; ++j) lsum[j] += rsum16(rs[j]);

        // C-layout P -> per-wave LDS -> A-layout fragments (same-wave LDS is
        // in-order on DScnt; no block barrier needed)
#pragma unroll
        for (int t = 0; t < 4; ++t)
#pragma unroll
            for (int j = 0; j < 8; ++j)
                lP[w][j + 8 * half][t * 16 + l16] = (__bf16)s[t][j];
        v16bf aP[2];
#pragma unroll
        for (int c = 0; c < 2; ++c) {
            const __bf16* p = &lP[w][l16][c * 32 + half * 16];
            aP[c] = frag_from2(p, p + 8);
        }

        // O += P @ V : K-depth 64 (2 chunks) x 4 dh tiles
#pragma unroll
        for (int t = 0; t < 4; ++t) {
            const __bf16* p = &lVt[cur][t * 16 + l16][half * 16];
            v16bf f0 = frag_from2(p, p + 8);
            v16bf f1 = frag_from2(p + 32, p + 40);
            o[t] = wmma_bf16(aP[0], f0, o[t]);
            o[t] = wmma_bf16(aP[1], f1, o[t]);
        }

        wait_async0();
        __syncthreads();
    }

    // write attn = O / l  -> [B, SQ, DV] at head offset
#pragma unroll
    for (int t = 0; t < 4; ++t)
#pragma unroll
        for (int j = 0; j < 8; ++j) {
            const int row = q0 + j + 8 * half;
            const size_t idx = ((size_t)(b * SQL + row)) * DVL + h * DH + t * 16 + l16;
            attn[idx] = o[t][j] / lsum[j];
        }
}

// ---------------------------------------------------------------------------
// ln_kernel: row layernorm over 1024 features. ADD: x = X + Y first.
// Wave reductions fully on the VALU (permlane16 / permlanex16).
// ---------------------------------------------------------------------------
template <bool ADD>
__global__ __launch_bounds__(256) void ln_kernel(
    const float* __restrict__ X,
    const float* __restrict__ Y,
    const float* __restrict__ g,
    const float* __restrict__ be,
    float*  __restrict__ outF,
    __bf16* __restrict__ outB)
{
    __shared__ float sred[8];
    const int row = blockIdx.x;
    const int tid = threadIdx.x;
    const int w = tid >> 5, lane = tid & 31;
    const size_t base = (size_t)row * DVL;

    float x[4];
    float s = 0.f;
#pragma unroll
    for (int i = 0; i < 4; ++i) {
        const int c = tid + i * 256;
        float vv = X[base + c];
        if (ADD) vv += Y[base + c];
        x[i] = vv;
        s += vv;
    }
    s = wave_sum32(s);
    if (lane == 0) sred[w] = s;
    __syncthreads();
    float tot = 0.f;
#pragma unroll
    for (int i = 0; i < 8; ++i) tot += sred[i];
    const float mu = tot * (1.f / DVL);
    __syncthreads();

    float sq = 0.f;
#pragma unroll
    for (int i = 0; i < 4; ++i) { float d = x[i] - mu; sq += d * d; }
    sq = wave_sum32(sq);
    if (lane == 0) sred[w] = sq;
    __syncthreads();
    tot = 0.f;
#pragma unroll
    for (int i = 0; i < 8; ++i) tot += sred[i];
    const float rstd = rsqrtf(tot * (1.f / DVL) + 1e-5f);

#pragma unroll
    for (int i = 0; i < 4; ++i) {
        const int c = tid + i * 256;
        float vv = (x[i] - mu) * rstd * g[c] + be[c];
        if (outF) outF[base + c] = vv;
        if (outB) outB[base + c] = (__bf16)vv;
    }
}

// ---------------------------------------------------------------------------
// Host launcher.  Workspace layout (168 MB total):
//   [  0,  32MB) q_f32 (reused later for t = y + relu(y@Wo+bo))
//   [ 32,  64MB) attn_f32
//   [ 64,  80MB) q_bf16  (pre-scaled by 1/sqrt(1024))
//   [ 80,  96MB) k_bf16
//   [ 96, 112MB) vT_bf16 [B,H,dh,SK]
//   [112, 128MB) y_bf16
//   [128, 144MB) Qin_bf16
//   [144, 160MB) Kin_bf16
//   [160, 168MB) Wt x4 (2MB each, transposed bf16 weights)
// ---------------------------------------------------------------------------
extern "C" void kernel_launch(void* const* d_in, const int* in_sizes, int n_in,
                              void* d_out, int out_size, void* d_ws, size_t ws_size,
                              hipStream_t stream) {
    (void)in_sizes; (void)n_in; (void)out_size; (void)ws_size;

    const float* Q  = (const float*)d_in[0];
    const float* K  = (const float*)d_in[1];
    const float* Wq = (const float*)d_in[2];
    const float* bq = (const float*)d_in[3];
    const float* Wk = (const float*)d_in[4];
    const float* bk = (const float*)d_in[5];
    const float* Wv = (const float*)d_in[6];
    const float* bv = (const float*)d_in[7];
    const float* Wo = (const float*)d_in[8];
    const float* bo = (const float*)d_in[9];
    const float* g0 = (const float*)d_in[10];
    const float* b0 = (const float*)d_in[11];
    const float* g1 = (const float*)d_in[12];
    const float* b1 = (const float*)d_in[13];
    float* out = (float*)d_out;

    const size_t MB = 1024 * 1024;
    char* ws = (char*)d_ws;
    float*  qf   = (float*) (ws + 0);
    float*  atf  = (float*) (ws + 32  * MB);
    __bf16* qb   = (__bf16*)(ws + 64  * MB);
    __bf16* kbf  = (__bf16*)(ws + 80  * MB);
    __bf16* vbT  = (__bf16*)(ws + 96  * MB);
    __bf16* yb   = (__bf16*)(ws + 112 * MB);
    __bf16* Qinb = (__bf16*)(ws + 128 * MB);
    __bf16* Kinb = (__bf16*)(ws + 144 * MB);
    __bf16* Wt0  = (__bf16*)(ws + 160 * MB);
    __bf16* Wt1  = (__bf16*)(ws + 162 * MB);
    __bf16* Wt2  = (__bf16*)(ws + 164 * MB);
    __bf16* Wt3  = (__bf16*)(ws + 166 * MB);

    const dim3 gemmGrid(DVL / 128, MROWS / 128);   // (8, 64)
    const dim3 attnGrid(SQL / 128, HH, BB);        // (8, 16, 8)
    const dim3 twGrid(DVL / 64, DVL / 64);         // (16, 16)

    // 0: one-time precision / layout conversion
    cvt_bf16_kernel<<<2048, 256, 0, stream>>>(Q, Qinb);
    cvt_bf16_kernel<<<2048, 256, 0, stream>>>(K, Kinb);
    transpose_w_kernel<<<twGrid, 256, 0, stream>>>(Wq, Wt0);
    transpose_w_kernel<<<twGrid, 256, 0, stream>>>(Wk, Wt1);
    transpose_w_kernel<<<twGrid, 256, 0, stream>>>(Wv, Wt2);
    transpose_w_kernel<<<twGrid, 256, 0, stream>>>(Wo, Wt3);

    // 1-3: projections (q bf16 pre-scaled; v emitted pre-transposed)
    proj_kernel<0, false><<<gemmGrid, 256, 0, stream>>>(Qinb, Wt0, bq, nullptr, qf, qb, 0.03125f);
    proj_kernel<0, false><<<gemmGrid, 256, 0, stream>>>(Kinb, Wt1, bk, nullptr, nullptr, kbf, 1.0f);
    proj_kernel<0, true ><<<gemmGrid, 256, 0, stream>>>(Kinb, Wt2, bv, nullptr, nullptr, vbT, 1.0f);
    // 4: flash attention
    attn_kernel<<<attnGrid, 256, 0, stream>>>(qb, kbf, vbT, atf);
    // 5: y = LN(q + attn) -> bf16
    ln_kernel<true><<<MROWS, 256, 0, stream>>>(qf, atf, g0, b0, nullptr, yb);
    // 6: t = y + relu(y @ Wo + bo)   (t reuses the q_f32 buffer)
    proj_kernel<1, false><<<gemmGrid, 256, 0, stream>>>(yb, Wt3, bo, yb, qf, nullptr, 1.0f);
    // 7: out = LN(t)
    ln_kernel<false><<<MROWS, 256, 0, stream>>>(qf, nullptr, g1, b1, out, nullptr);
}
